// ConstellationNet_conv4_59184649339045
// MI455X (gfx1250) — matrix operand order repaired and checked by reading, not compiled, not run on hardware
//
#include <hip/hip_runtime.h>
#include <hip/hip_bf16.h>

#define DEV __device__ __forceinline__

typedef __attribute__((ext_vector_type(16))) _Float16 v16h;
typedef __attribute__((ext_vector_type(8)))  float    v8f;
typedef unsigned int u32x4 __attribute__((ext_vector_type(4)));
typedef int          i32x8 __attribute__((ext_vector_type(8)));
typedef int          i32x4 __attribute__((ext_vector_type(4)));

#define WMMA_F16(A_, B_, C_) \
  __builtin_amdgcn_wmma_f32_16x16x32_f16(false, (A_), false, (B_), (short)0, (C_), false, false)

// ---------------- helpers ----------------

DEV float nan2(float x) {                 // np.nan_to_num(nan=0.001, posinf=1, neginf=1)
  if (isnan(x)) return 0.001f;
  if (isinf(x)) return 1.0f;
  return x;
}

DEV unsigned pcg(unsigned v) {
  v = v * 747796405u + 2891336453u;
  unsigned w = ((v >> ((v >> 28) + 4u)) ^ v) * 277803737u;
  return (w >> 22) ^ w;
}
DEV float gaussf(unsigned seed) {         // Box-Muller from hash
  unsigned a = pcg(seed * 2u + 1u), b = pcg(seed * 2u + 2u);
  float u1 = (float)((a & 0xFFFFFFu) + 1u) * (1.0f / 16777217.0f);
  float u2 = (float)(b & 0xFFFFFFu) * (1.0f / 16777216.0f);
  return sqrtf(-2.0f * logf(u1)) * cosf(6.28318530718f * u2);
}

// A fragment (16x32 f16), ISA 7.12.2 16-bit A layout; ld = 64
DEV v16h load_a_f16(const _Float16* A, int row0, int k0, int lane) {
  int m = lane & 15, half = lane >> 4;
  const _Float16* base = A + (size_t)(row0 + m) * 64 + k0;
  v16h a;
#pragma unroll
  for (int e = 0; e < 16; ++e) {
    int j = e >> 1, lo = e & 1;
    int k = (j < 4) ? (half * 8 + 2 * j + lo) : (16 + half * 8 + 2 * (j - 4) + lo);
    a[e] = base[k];
  }
  return a;
}
DEV v16h load_a_f32(const float* A, int row0, int k0, int lane) {
  int m = lane & 15, half = lane >> 4;
  const float* base = A + (size_t)(row0 + m) * 64 + k0;
  v16h a;
#pragma unroll
  for (int e = 0; e < 16; ++e) {
    int j = e >> 1, lo = e & 1;
    int k = (j < 4) ? (half * 8 + 2 * j + lo) : (16 + half * 8 + 2 * (j - 4) + lo);
    a[e] = (_Float16)base[k];
  }
  return a;
}
// B fragment (32x16 f16) holding W^T: B(k, j) = W[col0+j][k0+k], W row-major 64x64 f32
DEV v16h load_b_wT(const float* W, int col0, int k0, int lane) {
  int n = lane & 15, half = lane >> 4;
  v16h b;
#pragma unroll
  for (int e = 0; e < 16; ++e) {
    int j = e >> 1, lo = e & 1;
    int k = (j < 4) ? (half * 8 + 2 * j + lo) : (16 + half * 8 + 2 * (j - 4) + lo);
    b[e] = (_Float16)W[(size_t)(col0 + n) * 64 + (k0 + k)];
  }
  return b;
}

DEV float conv3x3(const float* x, const float* w, float bias,
                  int b, int oc, int i, int j, int h) {
  float acc = bias;
#pragma unroll
  for (int ic = 0; ic < 3; ++ic)
#pragma unroll
    for (int dy = 0; dy < 3; ++dy)
#pragma unroll
      for (int dx = 0; dx < 3; ++dx) {
        int ii = i + dy - 1, jj = j + dx - 1;
        if (ii >= 0 && ii < h && jj >= 0 && jj < h)
          acc += x[((size_t)(b * 3 + ic) * h + ii) * h + jj] *
                 w[((oc * 3 + ic) * 3 + dy) * 3 + dx];
      }
  return acc;
}

// TDM: DMA a contiguous run of `nelems` f16 elements from global into LDS (ISA sec 8.3/8.4).
// Degenerate 2D tile: tile_dim0 = tensor_dim0 = stride = nelems, tile_dim1 = tensor_dim1 = 1.
// This toolchain exposes the 6-arg builtin: (g0 u32x4, g1 i32x8, g2 i32x4, g3 i32x4, i32x8, cpol).
#if __has_builtin(__builtin_amdgcn_tensor_load_to_lds)
#define HAVE_TDM 1
DEV void tdm_load_f16(unsigned lds_off, const void* gptr, unsigned nelems) {
  unsigned long long ga = (unsigned long long)(size_t)gptr;
  u32x4 g0;
  g0[0] = 1u;                                     // count=1, user mode, no gather
  g0[1] = lds_off;                                // lds_addr [63:32]
  g0[2] = (unsigned)(ga & 0xFFFFFFFFu);           // global_addr[31:0]
  g0[3] = (unsigned)((ga >> 32) & 0x01FFFFFFu) | 0x80000000u; // ga[56:32] | type=2 @[127:126]
  i32x8 g1;
  g1[0] = 0x00010000;                             // wg_mask=0, data_size=1 (2 bytes)
  g1[1] = (int)((nelems & 0xFFFFu) << 16);        // tensor_dim0[15:0] @ [31:16]
  g1[2] = (int)((nelems >> 16) & 0xFFFFu) | (1 << 16); // tensor_dim0[31:16] | tensor_dim1=1
  g1[3] = (int)((nelems & 0xFFFFu) << 16);        // tensor_dim1 hi=0 | tile_dim0 @ [31:16]
  g1[4] = 1;                                      // tile_dim1 = 1, tile_dim2 = 0
  g1[5] = (int)nelems;                            // tensor_dim0_stride[31:0]
  g1[6] = 0;                                      // stride hi, dim1_stride lo
  g1[7] = 0;
  i32x4 z4 = {0, 0, 0, 0};
  i32x8 z8 = {0, 0, 0, 0, 0, 0, 0, 0};
  __builtin_amdgcn_tensor_load_to_lds(g0, g1, z4, z4, z8, 0);
}
#else
#define HAVE_TDM 0
#endif

// ---------------- kernels ----------------

__global__ void k_zero(float* p, long n) {
  long i = (long)blockIdx.x * blockDim.x + threadIdx.x;
  if (i < n) p[i] = 0.0f;
}

// pass 1: conv -> per-channel sum/sumsq (block = h*h threads, grid = B*64)
__global__ void k_conv_stats(const float* x, const float* w, const float* cb,
                             float* sum, float* ssq, int h) {
  extern __shared__ float sm[];
  int b = blockIdx.x >> 6, oc = blockIdx.x & 63;
  int tid = threadIdx.x, i = tid / h, j = tid % h;
  float val = conv3x3(x, w, cb[oc], b, oc, i, j, h);
  sm[tid] = val; sm[blockDim.x + tid] = val * val;
  __syncthreads();
  for (int s = blockDim.x >> 1; s > 0; s >>= 1) {
    if (tid < s) { sm[tid] += sm[tid + s]; sm[blockDim.x + tid] += sm[blockDim.x + tid + s]; }
    __syncthreads();
  }
  if (tid == 0) { atomicAdd(&sum[oc], sm[0]); atomicAdd(&ssq[oc], sm[blockDim.x]); }
}

// pass 2: conv -> BN -> ReLU -> maxpool2 -> cfm(f32), feats(f16, NxC), rn=|feat|^2
__global__ void k_conv_pool(const float* x, const float* w, const float* cb,
                            const float* sum, const float* ssq,
                            const float* g, const float* bb,
                            float* cfm, _Float16* feats, float* rn, int h) {
  int h2 = h >> 1, hw2 = h2 * h2;
  int b = blockIdx.x >> 6, oc = blockIdx.x & 63;
  int tid = threadIdx.x, y2 = tid / h2, x2 = tid % h2;
  float cnt = (float)(512 * h * h);
  float mean = sum[oc] / cnt;
  float var = ssq[oc] / cnt - mean * mean;
  float sc = rsqrtf(var + 1e-5f) * g[oc];
  float best = -1e30f;
#pragma unroll
  for (int dy = 0; dy < 2; ++dy)
#pragma unroll
    for (int dx = 0; dx < 2; ++dx) {
      float v0 = conv3x3(x, w, cb[oc], b, oc, 2 * y2 + dy, 2 * x2 + dx, h);
      best = fmaxf(best, (v0 - mean) * sc + bb[oc]);
    }
  float r = fmaxf(best, 0.0f);
  cfm[((size_t)(b * 64 + oc) * h2 + y2) * h2 + x2] = r;
  int n = b * hw2 + y2 * h2 + x2;
  feats[(size_t)n * 64 + oc] = (_Float16)r;
  atomicAdd(&rn[n], r * r);
}

__global__ void k_cluster_init(float* v, float* s, float* colsum, float* cn, int iter) {
  __shared__ float sm[64];
  int k = blockIdx.x, c = threadIdx.x;
  float g = gaussf((unsigned)(iter + 1) * 2654435769u + (unsigned)(k * 64 + c));
  v[k * 64 + c] = g;
  sm[c] = g * g;
  __syncthreads();
  for (int st = 32; st > 0; st >>= 1) { if (c < st) sm[c] += sm[c + st]; __syncthreads(); }
  if (c == 0) { cn[k] = sm[0]; s[k] = 0.0f; colsum[k] = 0.0f; }
}

// G = feats(Nx64,f16) @ v^T (WMMA) -> cd = sqrt(max(rn+cn-2G,0)); colsum += rowsoftmax(exp(-cd))
__global__ void k_cluster_gemm(const _Float16* feats, const float* v, const float* rn,
                               const float* cn, float* cd, float* colsum, int N) {
  __shared__ float tile[8][16][64];
  __shared__ float rsArr[8][16];
  int wave = threadIdx.x >> 5, lane = threadIdx.x & 31;
  int row0 = (blockIdx.x * 8 + wave) * 16;
  if (row0 >= N) return;
  if (row0 + 128 < N)   // prefetch next block's A stripe (global_prefetch_b8)
    __builtin_prefetch(feats + (size_t)(row0 + 128) * 64, 0, 1);
  v16h a0 = load_a_f16(feats, row0, 0, lane);
  v16h a1 = load_a_f16(feats, row0, 32, lane);
  int mbase = (lane >> 4) * 8, nb = lane & 15;
#pragma unroll
  for (int ct = 0; ct < 4; ++ct) {
    v16h b0 = load_b_wT(v, ct * 16, 0, lane);
    v16h b1 = load_b_wT(v, ct * 16, 32, lane);
    v8f c = {};
    c = WMMA_F16(a0, b0, c);
    c = WMMA_F16(a1, b1, c);
    int col = ct * 16 + nb;
    float cnv = cn[col];
#pragma unroll
    for (int r = 0; r < 8; ++r) {
      int ml = mbase + r, row = row0 + ml;
      float d2 = rn[row] + cnv - 2.0f * c[r];
      float cdv = sqrtf(fmaxf(d2, 0.0f));
      cd[(size_t)row * 64 + col] = cdv;
      tile[wave][ml][col] = expf(-cdv);   // BETA = 1
    }
  }
  __builtin_amdgcn_wave_barrier();
  if (lane < 16) {
    float rs = 0.0f;
    for (int k2 = 0; k2 < 64; ++k2) rs += tile[wave][lane][k2];
    rsArr[wave][lane] = rs;
  }
  __builtin_amdgcn_wave_barrier();
  int c0 = lane * 2, c1 = c0 + 1;
  float cs0 = 0.0f, cs1 = 0.0f;
  for (int rr = 0; rr < 16; ++rr) {
    float rs = rsArr[wave][rr];
    cs0 += nan2(tile[wave][rr][c0] / rs);
    cs1 += nan2(tile[wave][rr][c1] / rs);
  }
  atomicAdd(&colsum[c0], cs0);
  atomicAdd(&colsum[c1], cs1);
}

__global__ void k_cluster_update(const _Float16* feats, float* v, float* s,
                                 float* colsum, float* cn) {
  int k = threadIdx.x;                    // one block, 64 threads
  float cs = 0.0f;
  for (int c = 0; c < 64; ++c) cs += (float)feats[k * 64 + c];
  float col = colsum[k];
  float vp = nan2(col * cs / col);        // faithful to source (== cs unless col==0/inf)
  float delta = col;
  float mu = nan2(1.0f / (s[k] + delta)); // LBDA = 1
  float sq = 0.0f;
  for (int c = 0; c < 64; ++c) {
    float nv = nan2((1.0f - mu) * v[k * 64 + c] + mu * vp);
    v[k * 64 + c] = nv;
    sq += nv * nv;
  }
  cn[k] = sq;
  s[k] = nan2(s[k] + delta);
  __syncthreads();
  colsum[k] = 0.0f;
}

// f1 = cd + positional encoding, as f16 (N x 64)
__global__ void k_f1build(const float* cd, _Float16* f1, int h2, long total) {
  long idx = (long)blockIdx.x * blockDim.x + threadIdx.x;
  if (idx >= total) return;
  int k = (int)(idx & 63);
  long n = idx >> 6;
  int hw = h2 * h2, p = (int)(n % hw), y = p / h2, xx = p % h2;
  int t = k & 15;
  float pos = (k < 32) ? (float)y : (float)xx;
  float dv = expf((float)(2 * t) * (-9.210340371976184f / 32.0f)); // ln(1e4)/c, c=32
  float ang = pos * dv;
  float pe = (k & 16) ? cosf(ang) : sinf(ang);
  f1[idx] = (_Float16)(cd[idx] + pe);
}

// out(N x 64) = A(N x 64) @ W^T + bias ; outMode 0: f16 row-major, 1: f32 scatter (B,64,hw)
__global__ void k_proj(const void* Aptr, int aIsF32, const float* W, const float* bias,
                       void* outPtr, int outMode, int hw, int N) {
  int wave = threadIdx.x >> 5, lane = threadIdx.x & 31;
  int row0 = (blockIdx.x * 8 + wave) * 16;
  if (row0 >= N) return;
  if (row0 + 128 < N)   // prefetch next block's A stripe
    __builtin_prefetch((const char*)Aptr + (size_t)(row0 + 128) * 64 * (aIsF32 ? 4 : 2), 0, 1);
  v16h a0, a1;
  if (aIsF32) {
    a0 = load_a_f32((const float*)Aptr, row0, 0, lane);
    a1 = load_a_f32((const float*)Aptr, row0, 32, lane);
  } else {
    a0 = load_a_f16((const _Float16*)Aptr, row0, 0, lane);
    a1 = load_a_f16((const _Float16*)Aptr, row0, 32, lane);
  }
  int mbase = (lane >> 4) * 8, nb = lane & 15;
#pragma unroll
  for (int ct = 0; ct < 4; ++ct) {
    v16h b0 = load_b_wT(W, ct * 16, 0, lane);
    v16h b1 = load_b_wT(W, ct * 16, 32, lane);
    v8f c = {};
    c = WMMA_F16(a0, b0, c);
    c = WMMA_F16(a1, b1, c);
    int col = ct * 16 + nb;
    float bv = bias[col];
#pragma unroll
    for (int r = 0; r < 8; ++r) {
      int row = row0 + mbase + r;
      float val = c[r] + bv;
      if (outMode == 0) {
        ((_Float16*)outPtr)[(size_t)row * 64 + col] = (_Float16)val;
      } else {
        int bi = row / hw, pp = row % hw;
        ((float*)outPtr)[((size_t)bi * 64 + col) * hw + pp] = val;
      }
    }
  }
}

__global__ void k_wsum(const float* w_w, float* wsum) {   // sum the 4 head blocks
  int idx = blockIdx.x * blockDim.x + threadIdx.x;        // 4096
  int k = idx >> 6, i = idx & 63;
  wsum[idx] = w_w[k * 256 + i] + w_w[k * 256 + 64 + i] +
              w_w[k * 256 + 128 + i] + w_w[k * 256 + 192 + i];
}

// per image: rowscale[n] = max(softmax_row) = 1/sum(exp(s - smax)); fa = rowscale * fv
// fk tile staged into LDS by the Tensor Data Mover (TENSORcnt tracked).
__global__ void k_rowscale_fa(const _Float16* fq, const _Float16* fk, const _Float16* fv,
                              _Float16* fa, int hw) {
  __shared__ _Float16 lk[256 * 64];
  int b = blockIdx.x, n = threadIdx.x;
  size_t base = (size_t)b * hw * 64;
#if HAVE_TDM
  if (threadIdx.x < 32) {                     // wave 0 issues the DMA (TDM ignores EXEC)
    unsigned lds_off = (unsigned)(size_t)(void*)lk;   // low 32 bits = LDS byte offset
    tdm_load_f16(lds_off, fk + base, (unsigned)(hw * 64));
    __builtin_amdgcn_s_wait_tensorcnt(0);
  }
#else
  for (int c = 0; c < 64; ++c) lk[n * 64 + c] = fk[base + (size_t)n * 64 + c];
#endif
  float q[64];
#pragma unroll
  for (int c = 0; c < 64; ++c) q[c] = (float)fq[base + (size_t)n * 64 + c];
  __syncthreads();
  float mx = -1e30f, sumv = 0.0f;
  for (int m = 0; m < hw; ++m) {
    float dot = 0.0f;
#pragma unroll
    for (int c = 0; c < 64; ++c) dot += q[c] * (float)lk[m * 64 + c];
    float sv = dot * 0.125f;               // / sqrt(K=64)
    if (sv > mx) { sumv = sumv * expf(mx - sv) + 1.0f; mx = sv; }
    else sumv += expf(sv - mx);
  }
  float scale = 1.0f / sumv;
  for (int c = 0; c < 64; ++c)
    fa[base + (size_t)n * 64 + c] = (_Float16)(scale * (float)fv[base + (size_t)n * 64 + c]);
}

// 1x1 conv over concat[cfm, faout] -> xpre + per-channel stats (grid=B*3, block=hw)
__global__ void k_catconv_stats(const float* cfm, const float* faout, const float* cw,
                                const float* cbias, float* xpre,
                                float* sum, float* ssq, int hw) {
  extern __shared__ float sm[];
  int b = blockIdx.x / 3, oc = blockIdx.x % 3;
  int p = threadIdx.x;
  float acc = cbias[oc];
  for (int ic = 0; ic < 64; ++ic) {
    acc += cfm[(size_t)(b * 64 + ic) * hw + p] * cw[oc * 128 + ic];
    acc += faout[(size_t)(b * 64 + ic) * hw + p] * cw[oc * 128 + 64 + ic];
  }
  xpre[(size_t)(b * 3 + oc) * hw + p] = acc;
  sm[p] = acc; sm[blockDim.x + p] = acc * acc;
  __syncthreads();
  for (int s = blockDim.x >> 1; s > 0; s >>= 1) {
    if (p < s) { sm[p] += sm[p + s]; sm[blockDim.x + p] += sm[blockDim.x + p + s]; }
    __syncthreads();
  }
  if (p == 0) { atomicAdd(&sum[oc], sm[0]); atomicAdd(&ssq[oc], sm[blockDim.x]); }
}

__global__ void k_catconv_norm(const float* xpre, const float* sum, const float* ssq,
                               const float* g, const float* bb, float* xout,
                               int hw, int total) {
  int idx = blockIdx.x * blockDim.x + threadIdx.x;
  if (idx >= total) return;
  int c = (idx / hw) % 3;
  float cnt = (float)(512 * hw);
  float mean = sum[c] / cnt, var = ssq[c] / cnt - mean * mean;
  float v0 = (xpre[idx] - mean) * rsqrtf(var + 1e-5f) * g[c] + bb[c];
  xout[idx] = fmaxf(v0, 0.0f);
}

__global__ void k_final(const float* x, const float* ew, const float* eb, float* out) {
  int idx = blockIdx.x * blockDim.x + threadIdx.x;  // 512*128
  int b = idx >> 7, o = idx & 127;
  float acc = eb[o];
#pragma unroll
  for (int c = 0; c < 3; ++c) {
    const float* px = x + (size_t)(b * 3 + c) * 4;  // final map is 2x2
    float avg = 0.25f * (px[0] + px[1] + px[2] + px[3]);
    acc += avg * ew[o * 3 + c];
  }
  out[idx] = acc;
}

// ---------------- host orchestration ----------------

extern "C" void kernel_launch(void* const* d_in, const int* in_sizes, int n_in,
                              void* d_out, int out_size, void* d_ws, size_t ws_size,
                              hipStream_t stream) {
  const float* x_in   = (const float*)d_in[0];
  const float* conv_w = (const float*)d_in[1];
  const float* conv_b = (const float*)d_in[2];
  const float* bn1_g  = (const float*)d_in[3];
  const float* bn1_b  = (const float*)d_in[4];
  const float* cat_w  = (const float*)d_in[5];
  const float* cat_b  = (const float*)d_in[6];
  const float* bn2_g  = (const float*)d_in[7];
  const float* bn2_b  = (const float*)d_in[8];
  const float* wq_w   = (const float*)d_in[9];
  const float* wq_b   = (const float*)d_in[10];
  const float* wk_w   = (const float*)d_in[11];
  const float* wk_b   = (const float*)d_in[12];
  const float* wv_w   = (const float*)d_in[13];
  const float* wv_b   = (const float*)d_in[14];
  const float* w_w    = (const float*)d_in[15];
  const float* w_b    = (const float*)d_in[16];
  const float* emb_w  = (const float*)d_in[17];
  const float* emb_b  = (const float*)d_in[18];

  const int Nmax = 512 * 16 * 16;   // 131072 (iteration 0)
  char* p = (char*)d_ws;
  auto carve = [&](size_t bytes) { char* r = p; p += (bytes + 255) & ~(size_t)255; return (void*)r; };

  float*     XBUF   = (float*)    carve((size_t)512 * 3 * 256 * 4);     // next-x (<= B,3,16,16)
  float*     CFM    = (float*)    carve((size_t)512 * 64 * 256 * 4);    // 33.5 MB
  _Float16*  FEATS  = (_Float16*) carve((size_t)Nmax * 64 * 2);         // 16.8 MB
  float*     RN     = (float*)    carve((size_t)Nmax * 4);
  float*     CD     = (float*)    carve((size_t)Nmax * 64 * 4);         // d_map; later aliased FAOUT
  _Float16*  F1     = (_Float16*) carve((size_t)Nmax * 64 * 2);         // later aliased FA
  _Float16*  FQ     = (_Float16*) carve((size_t)Nmax * 64 * 2);
  _Float16*  FK     = (_Float16*) carve((size_t)Nmax * 64 * 2);
  _Float16*  FV     = (_Float16*) carve((size_t)Nmax * 64 * 2);
  float*     XPRE   = (float*)    carve((size_t)512 * 3 * 256 * 4);
  float*     V      = (float*)    carve(64 * 64 * 4);
  float*     S      = (float*)    carve(64 * 4);
  float*     COLSUM = (float*)    carve(64 * 4);
  float*     CN     = (float*)    carve(64 * 4);
  float*     WSUM   = (float*)    carve(64 * 64 * 4);
  float*     STATS  = (float*)    carve(136 * 4);  // sum1[64] ssq1[64] sum2[4] ssq2[4]
  float* SUM1 = STATS, *SSQ1 = STATS + 64, *SUM2 = STATS + 128, *SSQ2 = STATS + 132;
  _Float16* FA = F1;            // alias: f1 dead after fq/fk projections
  float* FAOUT = CD;            // alias: cd dead after f1build + fv projection

  const float* xcur = x_in;
  int h = 32;
  for (int it = 0; it < 4; ++it) {
    int h2 = h >> 1, hw = h2 * h2, N = 512 * hw;

    k_zero<<<1, 256, 0, stream>>>(STATS, 136);
    k_zero<<<(N + 255) / 256, 256, 0, stream>>>(RN, N);
    k_conv_stats<<<512 * 64, h * h, 2 * h * h * 4, stream>>>(xcur, conv_w, conv_b, SUM1, SSQ1, h);
    k_conv_pool <<<512 * 64, hw, 0, stream>>>(xcur, conv_w, conv_b, SUM1, SSQ1,
                                              bn1_g, bn1_b, CFM, FEATS, RN, h);
    k_cluster_init<<<64, 64, 0, stream>>>(V, S, COLSUM, CN, it);
    for (int ep = 0; ep < 10; ++ep) {
      k_cluster_gemm<<<N / 128, 256, 0, stream>>>(FEATS, V, RN, CN, CD, COLSUM, N);
      k_cluster_update<<<1, 64, 0, stream>>>(FEATS, V, S, COLSUM, CN);
    }
    long tot = (long)N * 64;
    k_f1build<<<(int)((tot + 255) / 256), 256, 0, stream>>>(CD, F1, h2, tot);
    k_proj<<<N / 128, 256, 0, stream>>>(F1, 0, wq_w, wq_b, FQ, 0, hw, N);
    k_proj<<<N / 128, 256, 0, stream>>>(F1, 0, wk_w, wk_b, FK, 0, hw, N);
    k_proj<<<N / 128, 256, 0, stream>>>(CD, 1, wv_w, wv_b, FV, 0, hw, N);  // f1p = d_map
    k_wsum<<<16, 256, 0, stream>>>(w_w, WSUM);
    k_rowscale_fa<<<512, hw, 0, stream>>>(FQ, FK, FV, FA, hw);
    k_proj<<<N / 128, 256, 0, stream>>>(FA, 0, WSUM, w_b, FAOUT, 1, hw, N);
    k_catconv_stats<<<512 * 3, hw, 2 * hw * 4, stream>>>(CFM, FAOUT, cat_w, cat_b,
                                                         XPRE, SUM2, SSQ2, hw);
    k_catconv_norm<<<(512 * 3 * hw + 255) / 256, 256, 0, stream>>>(XPRE, SUM2, SSQ2,
                                                                   bn2_g, bn2_b, XBUF,
                                                                   hw, 512 * 3 * hw);
    xcur = XBUF;
    h = h2;
  }
  k_final<<<(512 * 128) / 256, 256, 0, stream>>>(XBUF, emb_w, emb_b, (float*)d_out);
}
